// CATransformer1_82094004895774
// MI455X (gfx1250) — matrix operand order repaired and checked by reading, not compiled, not run on hardware
//
#include <hip/hip_runtime.h>
#include <hip/hip_bf16.h>
#include <math.h>

typedef __bf16 bf16;
typedef __attribute__((ext_vector_type(8)))  bf16  v8bf;
typedef __attribute__((ext_vector_type(16))) bf16  v16bf;
typedef __attribute__((ext_vector_type(8)))  float v8f;

#define DIM   384
#define NH    8
#define HID   1536
#define BATCH 16
#define NPIX  4096          // 64*64 pixels per image
#define CH    48            // per-head channels
#define CHP   64            // per-head channels padded to K-multiple
#define QKROWS 768          // q (384) + k (384) rows per batch
#define LOGIT_MAX 4.6051701860f
#define EPS_LN    1e-5f
#define EPS_NORM  1e-12f

// ---------------- workspace layout (bytes) ----------------
constexpr size_t OFF_WQ  = 0;                                           // 1152x384 bf16
constexpr size_t OFF_WP  = OFF_WQ  + (size_t)1152 * 384 * 2;            // 384x384 bf16
constexpr size_t OFF_W1  = OFF_WP  + (size_t)384  * 384 * 2;            // 1536x384 bf16
constexpr size_t OFF_W2  = OFF_W1  + (size_t)1536 * 384 * 2;            // 384x1536 bf16
constexpr size_t OFF_XNT = OFF_W2  + (size_t)384  * 1536 * 2;           // [B*N][384] bf16 (xnT -> attnoutT -> xn2T)
constexpr size_t OFF_QK  = OFF_XNT + (size_t)BATCH * NPIX * DIM * 2;    // [B][768][N] bf16 ; reused as x2 f32 (same bytes)
constexpr size_t OFF_VT  = OFF_QK  + (size_t)BATCH * QKROWS * NPIX * 2; // [B*N][8*64] bf16, padded v
constexpr size_t OFF_NRM = OFF_VT  + (size_t)BATCH * NPIX * NH * CHP * 2; // [B][768] f32 row norms
constexpr size_t OFF_LOG = OFF_NRM + (size_t)BATCH * QKROWS * 4;        // [B][8][48][64] f32 logits
constexpr size_t OFF_ATT = OFF_LOG + (size_t)BATCH * NH * CH * CHP * 4; // [B][8][48][64] bf16 attn

// ---------------- helpers ----------------
__device__ __forceinline__ v8f vzero8() {
    v8f z = {0.f, 0.f, 0.f, 0.f, 0.f, 0.f, 0.f, 0.f};
    return z;
}

__device__ __forceinline__ float waveReduceSum(float v) {
    #pragma unroll
    for (int o = 16; o > 0; o >>= 1) v += __shfl_xor(v, o, 32);
    return v;
}
__device__ __forceinline__ float waveReduceMax(float v) {
    #pragma unroll
    for (int o = 16; o > 0; o >>= 1) v = fmaxf(v, __shfl_xor(v, o, 32));
    return v;
}

__device__ __forceinline__ v16bf combine16(v8bf lo, v8bf hi) {
    v16bf r;
    #pragma unroll
    for (int i = 0; i < 8; ++i) { r[i] = lo[i]; r[i + 8] = hi[i]; }
    return r;
}

// Per-lane operand fetch matching the CDNA5 16-bit WMMA VGPR layouts:
// A 16x32: ap pre-offset by lane-hi*8 -> halves at +0 / +16 elements.
// B 32x16: bp pre-offset by lane-hi*8 -> halves at +0 / +16 elements.
__device__ __forceinline__ v16bf loadA(const bf16* ap) {
    return combine16(*(const v8bf*)ap, *(const v8bf*)(ap + 16));
}
__device__ __forceinline__ v16bf loadB(const bf16* bp) {
    return combine16(*(const v8bf*)bp, *(const v8bf*)(bp + 16));
}

// D(16x16,f32) += A(16xK bf16, row-major lda) * B(Kx16 bf16, column-contiguous, col stride ldb).
// Simple form: one operand set in flight -> no spills; compiler software-pipelines across
// k iterations on its own (partial s_wait_loadcnt observed in Round 1).
__device__ __forceinline__ void wmma_loop(const bf16* __restrict__ A, const bf16* B,
                                          int lda, int ldb, int K, v8f& acc)
{
    const int lane = threadIdx.x & 31;
    const int hi   = lane >> 4;
    const int row  = lane & 15;
    const bf16* ap = A + (size_t)row * lda + hi * 8;
    const bf16* bp = B + (size_t)row * ldb + hi * 8;
    for (int k = 0; k < K; k += 32) {
        v16bf a = loadA(ap + k);
        v16bf b = loadB(bp + k);
        acc = __builtin_amdgcn_wmma_f32_16x16x32_bf16(false, a, false, b,
                                                      (short)0, acc, false, false);
    }
}

// Register-blocked: one A tile reused across 4 adjacent N tiles (16x64 output per wave;
// 2 A-loads + 8 B-loads per 4 WMMAs = 2.5 loads/WMMA). All 10 loads of a k-step are
// issued before the 4 WMMAs so the waits can retire incrementally; only one operand
// set is live at a time (~75 VGPRs total) to avoid the scratch spills seen with
// explicit double-buffering.
__device__ __forceinline__ void wmma_loop_n4(const bf16* __restrict__ A, const bf16* B,
                                             int lda, int ldb, int K, v8f acc[4])
{
    const int lane = threadIdx.x & 31;
    const int hi   = lane >> 4;
    const int row  = lane & 15;
    const bf16* ap = A + (size_t)row * lda + hi * 8;
    const bf16* bp = B + (size_t)row * ldb + hi * 8;
    const size_t bstride = (size_t)16 * ldb;
    for (int k = 0; k < K; k += 32) {
        v16bf a  = loadA(ap + k);
        v16bf b0 = loadB(bp + k);
        v16bf b1 = loadB(bp + bstride + k);
        v16bf b2 = loadB(bp + 2 * bstride + k);
        v16bf b3 = loadB(bp + 3 * bstride + k);
        acc[0] = __builtin_amdgcn_wmma_f32_16x16x32_bf16(false, a, false, b0,
                                                         (short)0, acc[0], false, false);
        acc[1] = __builtin_amdgcn_wmma_f32_16x16x32_bf16(false, a, false, b1,
                                                         (short)0, acc[1], false, false);
        acc[2] = __builtin_amdgcn_wmma_f32_16x16x32_bf16(false, a, false, b2,
                                                         (short)0, acc[2], false, false);
        acc[3] = __builtin_amdgcn_wmma_f32_16x16x32_bf16(false, a, false, b3,
                                                         (short)0, acc[3], false, false);
    }
}

// ---------------- elementwise kernels ----------------
__global__ void k_f32_to_bf16(const float* __restrict__ s, bf16* __restrict__ d, int n) {
    int i = blockIdx.x * blockDim.x + threadIdx.x;
    if (i < n) d[i] = (bf16)s[i];
}

__global__ void k_fill_zero_bf16(bf16* __restrict__ d, size_t n) {
    size_t i = (size_t)blockIdx.x * blockDim.x + threadIdx.x;
    if (i < n) d[i] = (bf16)0.f;
}

// LayerNorm over channels of [B][C][N] f32, emit pixel-major bf16 [B*N][C]
__global__ void k_ln_to_bf16T(const float* __restrict__ x, const float* __restrict__ g,
                              const float* __restrict__ be, bf16* __restrict__ xT)
{
    const int p  = blockIdx.x * blockDim.x + threadIdx.x; // global pixel
    const int b  = p >> 12, nn = p & 4095;
    const float* xp = x + (size_t)b * DIM * NPIX + nn;
    float s = 0.f, s2 = 0.f;
    for (int c = 0; c < DIM; ++c) {
        float v = xp[(size_t)c * NPIX];
        s += v; s2 += v * v;
    }
    const float m   = s * (1.f / DIM);
    const float var = s2 * (1.f / DIM) - m * m;
    const float r   = rsqrtf(var + EPS_LN);
    bf16* o = xT + (size_t)p * DIM;
    for (int c = 0; c < DIM; ++c)
        o[c] = (bf16)((xp[(size_t)c * NPIX] - m) * r * g[c] + be[c]);
}

// ---------------- QKV GEMM: [1152x384] x xnT -> qk (channel-major) / v (pixel-major, head-padded)
__global__ void k_gemm_qkv(const bf16* __restrict__ wq, const float* __restrict__ bq,
                           const bf16* __restrict__ xT, bf16* __restrict__ qk,
                           bf16* __restrict__ vT)
{
    const int wid  = (blockIdx.x * blockDim.x + threadIdx.x) >> 5; // b*72*64 waves
    const int nt4  = wid & 63;                                     // group of 4 n-tiles
    const int ot   = (wid >> 6) % 72;
    const int b    = wid / (72 * 64);
    const int lane = threadIdx.x & 31, hi = lane >> 4, row = lane & 15;

    v8f acc[4] = {vzero8(), vzero8(), vzero8(), vzero8()};
    wmma_loop_n4(wq + (size_t)ot * 16 * DIM,
                 xT + ((size_t)b * NPIX + nt4 * 64) * DIM, DIM, DIM, DIM, acc);

    const int obase = ot * 16 + hi * 8;       // 8 consecutive output channels per lane
    #pragma unroll
    for (int j = 0; j < 4; ++j) {
        const int n = nt4 * 64 + j * 16 + row;
        if (obase < 2 * DIM) {                // q,k rows: channel-major [768][N]
            bf16* dst = qk + ((size_t)b * QKROWS + obase) * NPIX + n;
            #pragma unroll
            for (int r = 0; r < 8; ++r)
                dst[(size_t)r * NPIX] = (bf16)(acc[j][r] + bq[obase + r]);
        } else {                              // v: pixel-major, per-head padded 48->64
            const int vc = obase - 2 * DIM;   // 8-aligned, never crosses a 48-boundary
            const int h = vc / CH, cp = vc % CH;
            v8bf pk;
            #pragma unroll
            for (int r = 0; r < 8; ++r) pk[r] = (bf16)(acc[j][r] + bq[obase + r]);
            *(v8bf*)(vT + ((size_t)b * NPIX + n) * (NH * CHP) + h * CHP + cp) = pk;
        }
    }
}

// ---------------- q/k row L2 norms ----------------
__global__ void k_rownorm(const bf16* __restrict__ qk, float* __restrict__ nrm)
{
    const int wid  = (blockIdx.x * blockDim.x + threadIdx.x) >> 5; // row 0..B*768-1
    const int lane = threadIdx.x & 31;
    const bf16* p  = qk + (size_t)wid * NPIX;
    float s = 0.f;
    for (int n = lane; n < NPIX; n += 32) {
        float v = (float)p[n];
        s += v * v;
    }
    s = waveReduceSum(s);
    if (lane == 0) nrm[wid] = sqrtf(s);
}

// ---------------- gram: per (b,h) q . k^T over K=N, scaled logits ----------------
__global__ void k_gram(const bf16* __restrict__ qk, const float* __restrict__ nrm,
                       const float* __restrict__ ls, float* __restrict__ logits)
{
    const int wid  = (blockIdx.x * blockDim.x + threadIdx.x) >> 5; // b*8*3*3 waves
    const int mt   = wid % 3;
    const int nt   = (wid / 3) % 3;
    const int h    = (wid / 9) % NH;
    const int b    = wid / 72;
    const int lane = threadIdx.x & 31, hi = lane >> 4, row = lane & 15;

    const bf16* base = qk + (size_t)b * QKROWS * NPIX;
    v8f acc = vzero8();
    wmma_loop(base + (size_t)(h * CH + mt * 16) * NPIX,          // q rows (A)
              base + (size_t)(DIM + h * CH + nt * 16) * NPIX,    // k rows (B columns)
              NPIX, NPIX, NPIX, acc);

    const float sc = expf(fminf(ls[h], LOGIT_MAX));
    const int c0 = mt * 16 + hi * 8;
    const int d  = nt * 16 + row;
    const float kd = fmaxf(nrm[b * QKROWS + DIM + h * CH + d], EPS_NORM);
    float* dst = logits + ((size_t)(b * NH + h) * CH + c0) * CHP + d;
    #pragma unroll
    for (int r = 0; r < 8; ++r) {
        const float qc = fmaxf(nrm[b * QKROWS + h * CH + c0 + r], EPS_NORM);
        dst[(size_t)r * CHP] = acc[r] * sc / (qc * kd);
    }
}

// ---------------- softmax over 48, emit zero-padded 48x64 bf16 ----------------
__global__ void k_softmax(const float* __restrict__ logits, bf16* __restrict__ attn)
{
    const int rid  = (blockIdx.x * blockDim.x + threadIdx.x) >> 5; // B*8*48 rows
    const int lane = threadIdx.x & 31;
    const float* src = logits + (size_t)rid * CHP;
    float v0 = (lane < CH)      ? src[lane]      : -3.0e38f;
    float v1 = (lane + 32 < CH) ? src[lane + 32] : -3.0e38f;
    float m  = waveReduceMax(fmaxf(v0, v1));
    float e0 = (lane < CH)      ? expf(v0 - m) : 0.f;
    float e1 = (lane + 32 < CH) ? expf(v1 - m) : 0.f;
    float s  = waveReduceSum(e0 + e1);
    const float inv = 1.f / s;
    bf16* dst = attn + (size_t)rid * CHP;
    dst[lane]      = (bf16)(e0 * inv);
    dst[lane + 32] = (bf16)(e1 * inv);   // cols 48..63 -> exact zeros (K padding)
}

// ---------------- attention output: attn(48x64) x v -> attnoutT (pixel-major) ----------------
__global__ void k_attn_out(const bf16* __restrict__ attn, const bf16* __restrict__ vT,
                           bf16* __restrict__ aoT)
{
    const int wid  = (blockIdx.x * blockDim.x + threadIdx.x) >> 5; // b*8*3*64
    const int nt4  = wid & 63;
    int t = wid >> 6;
    const int mt = t % 3; t /= 3;
    const int h  = t % NH;
    const int b  = t / NH;
    const int lane = threadIdx.x & 31, hi = lane >> 4, row = lane & 15;

    v8f acc[4] = {vzero8(), vzero8(), vzero8(), vzero8()};
    wmma_loop_n4(attn + ((size_t)(b * NH + h) * CH + mt * 16) * CHP,
                 vT + ((size_t)b * NPIX + nt4 * 64) * (NH * CHP) + h * CHP,
                 CHP, NH * CHP, CHP, acc);

    const int cp = mt * 16 + hi * 8;
    #pragma unroll
    for (int j = 0; j < 4; ++j) {
        const int n = nt4 * 64 + j * 16 + row;
        v8bf pk;
        #pragma unroll
        for (int r = 0; r < 8; ++r) pk[r] = (bf16)acc[j][r];
        *(v8bf*)(aoT + ((size_t)b * NPIX + n) * DIM + h * CH + cp) = pk;
    }
}

// ---------------- proj GEMM + residual: x2 = x + Wp*attnout + bp ----------------
__global__ void k_gemm_proj(const bf16* __restrict__ wp, const float* __restrict__ bp,
                            const bf16* __restrict__ aoT, const float* __restrict__ x,
                            float* __restrict__ x2)
{
    const int wid  = (blockIdx.x * blockDim.x + threadIdx.x) >> 5; // b*24*64
    const int nt4  = wid & 63;
    const int ot   = (wid >> 6) % 24;
    const int b    = wid / (24 * 64);
    const int lane = threadIdx.x & 31, hi = lane >> 4, row = lane & 15;

    v8f acc[4] = {vzero8(), vzero8(), vzero8(), vzero8()};
    wmma_loop_n4(wp + (size_t)ot * 16 * DIM,
                 aoT + ((size_t)b * NPIX + nt4 * 64) * DIM, DIM, DIM, DIM, acc);

    const int o0 = ot * 16 + hi * 8;
    #pragma unroll
    for (int j = 0; j < 4; ++j) {
        const int n = nt4 * 64 + j * 16 + row;
        const size_t base = ((size_t)b * DIM + o0) * NPIX + n;
        #pragma unroll
        for (int r = 0; r < 8; ++r)
            x2[base + (size_t)r * NPIX] = x[base + (size_t)r * NPIX] + acc[j][r] + bp[o0 + r];
    }
}

// ---------------- fused FFN: h=GELU(W1*xn2+b1) staged in LDS, out = x2 + W2*h + b2 ----------------
__global__ void __launch_bounds__(256)
k_ffn(const bf16* __restrict__ w1, const float* __restrict__ b1,
      const bf16* __restrict__ w2, const float* __restrict__ b2,
      const bf16* __restrict__ xT, const float* __restrict__ x2,
      float* __restrict__ out)
{
    __shared__ __align__(16) bf16 hbuf[16 * HID];  // 48 KB: hidden for 16 pixels, pixel-major
    const int p0 = blockIdx.x * 16;                // 16 pixels per workgroup
    const int b  = p0 >> 12, nn = p0 & 4095;
    const int wave = threadIdx.x >> 5;
    const int lane = threadIdx.x & 31, hi = lane >> 4, row = lane & 15;
    const bf16* Bsrc = xT + (size_t)p0 * DIM;

    // FFN1: 96 m-tiles of 16 across 8 waves (uniform per wave)
    for (int mt = wave; mt < HID / 16; mt += 8) {
        v8f acc = vzero8();
        wmma_loop(w1 + (size_t)mt * 16 * DIM, Bsrc, DIM, DIM, DIM, acc);
        const int o0 = mt * 16 + hi * 8;
        v8bf pk;
        #pragma unroll
        for (int r = 0; r < 8; ++r) {
            float v = acc[r] + b1[o0 + r];
            v = 0.5f * v * (1.f + erff(v * 0.70710678f));  // exact GELU
            pk[r] = (bf16)v;
        }
        *(v8bf*)(&hbuf[(size_t)row * HID + o0]) = pk;
    }
    __syncthreads();

    // FFN2: 24 m-tiles, K=1536 straight out of LDS
    for (int mt = wave; mt < DIM / 16; mt += 8) {
        v8f acc = vzero8();
        wmma_loop(w2 + (size_t)mt * 16 * HID, (const bf16*)hbuf, HID, HID, HID, acc);
        const int o0 = mt * 16 + hi * 8;
        const size_t base = ((size_t)b * DIM + o0) * NPIX + nn + row;
        #pragma unroll
        for (int r = 0; r < 8; ++r)
            out[base + (size_t)r * NPIX] = x2[base + (size_t)r * NPIX] + acc[r] + b2[o0 + r];
    }
}

// ---------------- host launch ----------------
extern "C" void kernel_launch(void* const* d_in, const int* in_sizes, int n_in,
                              void* d_out, int out_size, void* d_ws, size_t ws_size,
                              hipStream_t stream)
{
    const float* x   = (const float*)d_in[0];
    const float* g1  = (const float*)d_in[1];
    const float* be1 = (const float*)d_in[2];
    const float* wq  = (const float*)d_in[3];
    const float* bq  = (const float*)d_in[4];
    const float* ls  = (const float*)d_in[5];
    const float* wp  = (const float*)d_in[6];
    const float* bp  = (const float*)d_in[7];
    const float* g2  = (const float*)d_in[8];
    const float* be2 = (const float*)d_in[9];
    const float* w1  = (const float*)d_in[10];
    const float* b1  = (const float*)d_in[11];
    const float* w2  = (const float*)d_in[12];
    const float* b2  = (const float*)d_in[13];

    char* ws = (char*)d_ws;
    bf16*  wqB    = (bf16*)(ws + OFF_WQ);
    bf16*  wpB    = (bf16*)(ws + OFF_WP);
    bf16*  w1B    = (bf16*)(ws + OFF_W1);
    bf16*  w2B    = (bf16*)(ws + OFF_W2);
    bf16*  xnT    = (bf16*)(ws + OFF_XNT);   // later reused as attnoutT, then xn2T
    bf16*  qkB    = (bf16*)(ws + OFF_QK);
    float* x2     = (float*)(ws + OFF_QK);   // byte-exact overlay after qk is dead
    bf16*  vT     = (bf16*)(ws + OFF_VT);
    float* nrm    = (float*)(ws + OFF_NRM);
    float* logits = (float*)(ws + OFF_LOG);
    bf16*  attn   = (bf16*)(ws + OFF_ATT);

    // weight conversion f32 -> bf16
    k_f32_to_bf16<<<(1152 * 384 + 255) / 256, 256, 0, stream>>>(wq, wqB, 1152 * 384);
    k_f32_to_bf16<<<(384 * 384 + 255) / 256, 256, 0, stream>>>(wp, wpB, 384 * 384);
    k_f32_to_bf16<<<(1536 * 384 + 255) / 256, 256, 0, stream>>>(w1, w1B, 1536 * 384);
    k_f32_to_bf16<<<(384 * 1536 + 255) / 256, 256, 0, stream>>>(w2, w2B, 384 * 1536);

    // zero v (only the 48..63 padding columns matter, whole-buffer fill is simplest)
    {
        size_t nv = (size_t)BATCH * NPIX * NH * CHP;
        k_fill_zero_bf16<<<(unsigned)((nv + 255) / 256), 256, 0, stream>>>(vT, nv);
    }

    // LN1 -> pixel-major bf16
    k_ln_to_bf16T<<<BATCH * NPIX / 256, 256, 0, stream>>>(x, g1, be1, xnT);

    // QKV GEMM: 16*72*64 waves / 8 waves per block
    k_gemm_qkv<<<BATCH * 72 * 64 / 8, 256, 0, stream>>>(wqB, bq, xnT, qkB, vT);

    // row norms of q,k: 16*768 rows / 8 per block
    k_rownorm<<<BATCH * QKROWS / 8, 256, 0, stream>>>(qkB, nrm);

    // gram + scale: 16*8*9 waves / 8 per block
    k_gram<<<BATCH * NH * 9 / 8, 256, 0, stream>>>(qkB, nrm, ls, logits);

    // softmax: 16*8*48 rows / 8 per block
    k_softmax<<<BATCH * NH * CH / 8, 256, 0, stream>>>(logits, attn);

    // attn x v -> attnoutT (reuses xnT region)
    k_attn_out<<<BATCH * NH * 3 * 64 / 8, 256, 0, stream>>>(attn, vT, xnT);

    // proj + residual -> x2 (overlays dead qk region)
    k_gemm_proj<<<BATCH * 24 * 64 / 8, 256, 0, stream>>>(wpB, bp, xnT, x, x2);

    // LN2 -> pixel-major bf16 (overwrites attnoutT, which is now dead)
    k_ln_to_bf16T<<<BATCH * NPIX / 256, 256, 0, stream>>>(x2, g2, be2, xnT);

    // fused FFN -> final output
    k_ffn<<<BATCH * NPIX / 16, 256, 0, stream>>>(w1B, b1, w2B, b2, xnT, x2, (float*)d_out);
}
